// Motif_46901042872606
// MI455X (gfx1250) — matrix operand order repaired and checked by reading, not compiled, not run on hardware
//
#include <hip/hip_runtime.h>
#include <cmath>

// ---------------------------------------------------------------------------
// Problem constants (from reference)
// ---------------------------------------------------------------------------
#define N_NODES 32768   // N_H == N_T
#define N_EDGES 524288
#define N_G     512
#define N_BLK   4

typedef __attribute__((ext_vector_type(16))) __bf16 v16bf;
typedef __attribute__((ext_vector_type(8)))  __bf16 bf16x8;
typedef __attribute__((ext_vector_type(8)))  float  v8f;

// ---------------------------------------------------------------------------
// Device helpers
// ---------------------------------------------------------------------------
__device__ __forceinline__ void atomicMaxFloat(float* addr, float val) {
  // int/uint ordering trick; buffer must be initialized to -inf
  if (val >= 0.0f) atomicMax((int*)addr, __float_as_int(val));
  else             atomicMin((unsigned int*)addr, __float_as_uint(val));
}

// ---------------------------------------------------------------------------
// WMMA GEMM: Y[m, co+n] = sum_k X[m,k] * W[k,n] + bias[n]
// X: [M x DIN] row-major f32, W: [DIN x dout] row-major f32, Y row stride ldo.
// Workgroup = 4 waves = 64 rows x 64 cols; wave tile = 16 rows x 64 cols
// (4 f32 accumulators, A fragment reused 4x). W is staged transposed in LDS
// as bf16 so each B fragment is two contiguous ds_load_b128. A fragments are
// two contiguous 8-float runs -> four global_load_b128 + v_cvt_pk_bf16_f32.
// DIN is a template parameter so the K-loop fully unrolls (DIN/32 * 4 WMMAs).
// ---------------------------------------------------------------------------
template <int DIN>
__global__ void gemm_wmma_bf16(const float* __restrict__ X,
                               const float* __restrict__ W,
                               const float* __restrict__ Bias,
                               float* __restrict__ Y,
                               int M, int dout, int ldo, int co) {
  __shared__ __bf16 Wlds[64 * DIN];   // [n_local][k] (transposed), bf16
  const int tid  = threadIdx.x;
  const int lane = tid & 31;
  const int wave = tid >> 5;
  const int tileM = (blockIdx.x * 4 + wave) * 16;
  const int tileN = blockIdx.y * 64;

  // Cooperative transpose+convert of the 64-column W panel into LDS.
  for (int idx = tid; idx < 64 * DIN; idx += 128) {
    const int n = idx / DIN, kq = idx - n * DIN;
    Wlds[idx] = (__bf16)W[(size_t)kq * dout + tileN + n];
  }
  __syncthreads();

  const int half = lane >> 4;          // 0: lanes 0-15, 1: lanes 16-31
  const int rowA = tileM + (lane & 15);
  const int nloc = lane & 15;
  const float4* Xr = (const float4*)(X + (size_t)rowA * DIN);

  v8f acc[4] = {};
#pragma unroll
  for (int kk = 0; kk < DIN; kk += 32) {
    // A fragment (bf16 16x32 layout): lane holds K = kk+8h+[0..7] and
    // K = kk+16+8h+[0..7] -> two 8-float contiguous runs.
    const int base0 = (kk + 8 * half) >> 2;
    const int base1 = (kk + 16 + 8 * half) >> 2;
    const float4 xa0 = Xr[base0], xa1 = Xr[base0 + 1];
    const float4 xa2 = Xr[base1], xa3 = Xr[base1 + 1];
    v16bf a;
    a[0] = (__bf16)xa0.x; a[1] = (__bf16)xa0.y; a[2] = (__bf16)xa0.z; a[3] = (__bf16)xa0.w;
    a[4] = (__bf16)xa1.x; a[5] = (__bf16)xa1.y; a[6] = (__bf16)xa1.z; a[7] = (__bf16)xa1.w;
    a[8]  = (__bf16)xa2.x; a[9]  = (__bf16)xa2.y; a[10] = (__bf16)xa2.z; a[11] = (__bf16)xa2.w;
    a[12] = (__bf16)xa3.x; a[13] = (__bf16)xa3.y; a[14] = (__bf16)xa3.z; a[15] = (__bf16)xa3.w;
#pragma unroll
    for (int ng = 0; ng < 4; ++ng) {
      // B fragment (bf16 32x16): lane holds K = kk+8h+[0..7] and K = kk+16+8h+[0..7]
      const __bf16* wp = &Wlds[(ng * 16 + nloc) * DIN + kk + 8 * half];
      const bf16x8 b0 = *(const bf16x8*)(wp);
      const bf16x8 b1 = *(const bf16x8*)(wp + 16);
      v16bf b;
#pragma unroll
      for (int e = 0; e < 8; ++e) { b[e] = b0[e]; b[e + 8] = b1[e]; }
      acc[ng] = __builtin_amdgcn_wmma_f32_16x16x32_bf16(false, a, false, b,
                                                        (short)0, acc[ng], false, false);
    }
  }
  // C/D layout: VGPR vv, lane L -> row = vv + 8*(L/16), col = L%16
#pragma unroll
  for (int ng = 0; ng < 4; ++ng) {
    const float bv = Bias[tileN + ng * 16 + nloc];
#pragma unroll
    for (int vv = 0; vv < 8; ++vv) {
      const int m = tileM + vv + 8 * half;
      Y[(size_t)m * ldo + co + tileN + ng * 16 + nloc] = acc[ng][vv] + bv;
    }
  }
}

// ---------------------------------------------------------------------------
// Elementwise / segment kernels
// ---------------------------------------------------------------------------
__global__ void fill_f32(float* p, float v, int n) {
  int i = blockIdx.x * blockDim.x + threadIdx.x;
  if (i < n) p[i] = v;
}

__global__ void count_nodes(const int* __restrict__ batch, float* cnt, int n) {
  int i = blockIdx.x * blockDim.x + threadIdx.x;
  if (i < n) atomicAdd(&cnt[batch[i]], 1.0f);
}

// Graph-LayerNorm pass 1: per-graph sum / sumsq via atomics (float4 row loads)
__global__ void gln_stats(const float* __restrict__ x, const int* __restrict__ batch,
                          float* gsum, float* gss, int n, int C) {
  int i = blockIdx.x * blockDim.x + threadIdx.x;
  if (i >= n) return;
  const float4* xr = (const float4*)(x + (size_t)i * C);
  float s = 0.0f, ss = 0.0f;
  for (int c = 0; c < C / 4; ++c) {
    float4 v = xr[c];
    s  += v.x + v.y + v.z + v.w;
    ss += v.x * v.x + v.y * v.y + v.z * v.z + v.w * v.w;
  }
  int g = batch[i];
  atomicAdd(&gsum[g], s);
  atomicAdd(&gss[g], ss);
}

// Graph-LayerNorm pass 2 (+fused activation): mode 0=none 1=relu 2=elu
__global__ void gln_apply(float* x, const int* __restrict__ batch,
                          const float* __restrict__ gsum, const float* __restrict__ gss,
                          const float* __restrict__ cnt,
                          const float* __restrict__ w, const float* __restrict__ b,
                          int n, int C, int mode) {
  int idx = blockIdx.x * blockDim.x + threadIdx.x;
  if (idx >= n * C) return;
  int node = idx / C, c = idx - node * C;
  int g = batch[node];
  float cc   = fmaxf(cnt[g] * (float)C, 1.0f);
  float mean = gsum[g] / cc;
  float var  = gss[g] / cc - mean * mean;
  float y = (x[idx] - mean) * rsqrtf(var + 1e-5f) * w[c] + b[c];
  if (mode == 1)      y = fmaxf(y, 0.0f);
  else if (mode == 2) y = (y > 0.0f) ? y : (__expf(y) - 1.0f);
  x[idx] = y;
}

// Attention logits + fused per-dst segment max. H=2, C templated (32/64).
template <int C>
__global__ void edge_logit(const float* __restrict__ q, const float* __restrict__ k,
                           const int* __restrict__ src, const int* __restrict__ dst,
                           float* logit, float* mbuf, int E, float scale) {
  int i = blockIdx.x * blockDim.x + threadIdx.x;
  if (i >= E * 2) return;
  int e = i >> 1, h = i & 1;
  int sN = src[e], dN = dst[e];
  const float4* qp = (const float4*)(q + (size_t)dN * (2 * C) + h * C);
  const float4* kp = (const float4*)(k + (size_t)sN * (2 * C) + h * C);
  float acc = 0.0f;
#pragma unroll
  for (int c = 0; c < C / 4; ++c) {
    float4 qa = qp[c], ka = kp[c];
    acc += qa.x * ka.x + qa.y * ka.y + qa.z * ka.z + qa.w * ka.w;
  }
  acc *= scale;
  logit[i] = acc;
  atomicMaxFloat(&mbuf[dN * 2 + h], acc);
}

// exp(logit - max) in-place + denominator accumulation (H=2)
__global__ void edge_exp(float* logit, const int* __restrict__ dst,
                         const float* __restrict__ mbuf, float* den, int E) {
  int i = blockIdx.x * blockDim.x + threadIdx.x;
  if (i >= E * 2) return;
  int e = i >> 1, h = i & 1;
  int dN = dst[e];
  float mm = mbuf[dN * 2 + h];
  if (!(mm >= -3.0e38f && mm <= 3.0e38f)) mm = 0.0f;  // non-finite -> 0 (ref semantics)
  float ex = __expf(logit[i] - mm);
  logit[i] = ex;
  atomicAdd(&den[dN * 2 + h], ex);
}

// out[dst] += alpha * v[src]  (skip projection pre-stored in out). H=2.
template <int C>
__global__ void edge_accum(const float* __restrict__ ex, const float* __restrict__ den,
                           const float* __restrict__ v,
                           const int* __restrict__ src, const int* __restrict__ dst,
                           float* out, int E, int ldo, int co) {
  int i = blockIdx.x * blockDim.x + threadIdx.x;
  if (i >= E * 2) return;
  int e = i >> 1, h = i & 1;
  int sN = src[e], dN = dst[e];
  float alpha = ex[i] / (den[dN * 2 + h] + 1e-16f);
  const float4* vp = (const float4*)(v + (size_t)sN * (2 * C) + h * C);
  float* op = out + (size_t)dN * ldo + co + h * C;
#pragma unroll
  for (int c = 0; c < C / 4; ++c) {
    float4 va = vp[c];
    atomicAdd(&op[4 * c + 0], alpha * va.x);
    atomicAdd(&op[4 * c + 1], alpha * va.y);
    atomicAdd(&op[4 * c + 2], alpha * va.z);
    atomicAdd(&op[4 * c + 3], alpha * va.w);
  }
}

// ---------------------------------------------------------------------------
// SAGPool(min_score=-1) -> softmax score -> global_add_pool
// ---------------------------------------------------------------------------
__global__ void sag_score(const float* __restrict__ x, const float* __restrict__ Wrel,
                          const float* __restrict__ Wroot, float* s1, float* s2, int n) {
  int i = blockIdx.x * blockDim.x + threadIdx.x;
  if (i >= n) return;
  const float4* xr = (const float4*)(x + (size_t)i * 128);
  const float4* wr = (const float4*)Wrel;
  const float4* wo = (const float4*)Wroot;
  float a = 0.0f, b = 0.0f;
#pragma unroll
  for (int c = 0; c < 32; ++c) {
    float4 xv = xr[c], rv = wr[c], ov = wo[c];
    a += xv.x * rv.x + xv.y * rv.y + xv.z * rv.z + xv.w * rv.w;
    b += xv.x * ov.x + xv.y * ov.y + xv.z * ov.z + xv.w * ov.w;
  }
  s1[i] = a; s2[i] = b;
}

__global__ void sag_agg(const float* __restrict__ s1, const int* __restrict__ src,
                        const int* __restrict__ dst, float* agg, int E) {
  int e = blockIdx.x * blockDim.x + threadIdx.x;
  if (e < E) atomicAdd(&agg[dst[e]], s1[src[e]]);
}

__global__ void sag_s(float* s, const float* __restrict__ agg, const float* __restrict__ s2,
                      const float* __restrict__ brel, const int* __restrict__ batch,
                      float* gm, int n) {
  int i = blockIdx.x * blockDim.x + threadIdx.x;
  if (i >= n) return;
  float v = agg[i] + brel[0] + s2[i];
  s[i] = v;
  atomicMaxFloat(&gm[batch[i]], v);
}

__global__ void sag_exp(float* s, const int* __restrict__ batch,
                        const float* __restrict__ gm, float* gden, int n) {
  int i = blockIdx.x * blockDim.x + threadIdx.x;
  if (i >= n) return;
  float mm = gm[batch[i]];
  if (!(mm >= -3.0e38f && mm <= 3.0e38f)) mm = 0.0f;
  float ex = __expf(s[i] - mm);
  s[i] = ex;
  atomicAdd(&gden[batch[i]], ex);
}

__global__ void sag_emb(const float* __restrict__ x, const float* __restrict__ s,
                        const float* __restrict__ gden, const int* __restrict__ batch,
                        float* out, int n, int blk) {
  int idx = blockIdx.x * blockDim.x + threadIdx.x;
  if (idx >= n * 128) return;
  int node = idx >> 7, c = idx & 127;
  int g = batch[node];
  float score = s[node] / (gden[g] + 1e-16f);
  atomicAdd(&out[(size_t)g * (N_BLK * 128) + blk * 128 + c],
            x[(size_t)node * 128 + c] * score);
}

// ---------------------------------------------------------------------------
// Host orchestration
// ---------------------------------------------------------------------------
static inline void launch_fill(hipStream_t s, float* p, float v, int n) {
  fill_f32<<<(n + 255) / 256, 256, 0, s>>>(p, v, n);
}
static inline void launch_gemm(hipStream_t s, const float* X, const float* W, const float* B,
                               float* Y, int M, int din, int dout, int ldo, int co) {
  dim3 grid(M / 64, dout / 64);
  if (din == 64)
    gemm_wmma_bf16<64><<<grid, 128, 0, s>>>(X, W, B, Y, M, dout, ldo, co);
  else
    gemm_wmma_bf16<128><<<grid, 128, 0, s>>>(X, W, B, Y, M, dout, ldo, co);
}

struct TCP { const float *Wq, *bq, *Wk, *bk, *Wv, *bv, *Ws, *bs; };

// TransformerConv: out (pre-loaded with x_dst@Ws+bs) += softmax-attention(v)
static void run_tconv(hipStream_t s,
                      const float* xsrc, const float* xdst, int din,
                      const int* srcI, const int* dstI, int nsrc, int ndst,
                      const TCP& p, int C,   // H == 2
                      float* out, int ldo, int co,
                      float* qb, float* kb, float* vb,
                      float* elog, float* mb, float* db) {
  const int HC = 2 * C;
  launch_gemm(s, xdst, p.Wq, p.bq, qb, ndst, din, HC, HC, 0);
  launch_gemm(s, xsrc, p.Wk, p.bk, kb, nsrc, din, HC, HC, 0);
  launch_gemm(s, xsrc, p.Wv, p.bv, vb, nsrc, din, HC, HC, 0);
  launch_gemm(s, xdst, p.Ws, p.bs, out, ndst, din, HC, ldo, co);   // root skip
  launch_fill(s, mb, -INFINITY, ndst * 2);
  launch_fill(s, db, 0.0f, ndst * 2);
  const int EH = N_EDGES * 2;
  const float scale = rsqrtf((float)C);
  if (C == 64) {
    edge_logit<64><<<(EH + 255) / 256, 256, 0, s>>>(qb, kb, srcI, dstI, elog, mb, N_EDGES, scale);
    edge_exp<<<(EH + 255) / 256, 256, 0, s>>>(elog, dstI, mb, db, N_EDGES);
    edge_accum<64><<<(EH + 255) / 256, 256, 0, s>>>(elog, db, vb, srcI, dstI, out, N_EDGES, ldo, co);
  } else {
    edge_logit<32><<<(EH + 255) / 256, 256, 0, s>>>(qb, kb, srcI, dstI, elog, mb, N_EDGES, scale);
    edge_exp<<<(EH + 255) / 256, 256, 0, s>>>(elog, dstI, mb, db, N_EDGES);
    edge_accum<32><<<(EH + 255) / 256, 256, 0, s>>>(elog, db, vb, srcI, dstI, out, N_EDGES, ldo, co);
  }
}

static void run_gln(hipStream_t s, float* x, const int* batch, const float* cnt,
                    const float* w, const float* b, int n, int C, int mode,
                    float* gsum, float* gss) {
  launch_fill(s, gsum, 0.0f, N_G);
  launch_fill(s, gss, 0.0f, N_G);
  gln_stats<<<(n + 255) / 256, 256, 0, s>>>(x, batch, gsum, gss, n, C);
  gln_apply<<<(n * C + 255) / 256, 256, 0, s>>>(x, batch, gsum, gss, cnt, w, b, n, C, mode);
}

static void run_sag(hipStream_t s, const float* x, const int* srcI, const int* dstI,
                    const int* batch, const float* Wrel, const float* Wroot,
                    const float* brel, float* out, int n, int blk,
                    float* s1, float* s2, float* agg, float* gm, float* gden) {
  launch_fill(s, agg, 0.0f, n);
  launch_fill(s, gm, -INFINITY, N_G);
  launch_fill(s, gden, 0.0f, N_G);
  sag_score<<<(n + 255) / 256, 256, 0, s>>>(x, Wrel, Wroot, s1, s2, n);
  sag_agg<<<(N_EDGES + 255) / 256, 256, 0, s>>>(s1, srcI, dstI, agg, N_EDGES);
  sag_s<<<(n + 255) / 256, 256, 0, s>>>(s1, agg, s2, brel, batch, gm, n);  // s1 <- s
  sag_exp<<<(n + 255) / 256, 256, 0, s>>>(s1, batch, gm, gden, n);         // s1 <- exp
  sag_emb<<<(n * 128 + 255) / 256, 256, 0, s>>>(x, s1, gden, batch, out, n, blk);
}

extern "C" void kernel_launch(void* const* d_in, const int* in_sizes, int n_in,
                              void* d_out, int out_size, void* d_ws, size_t ws_size,
                              hipStream_t stream) {
  (void)in_sizes; (void)n_in; (void)ws_size;
  const float* h_x_in = (const float*)d_in[0];
  const float* t_x_in = (const float*)d_in[1];
  const int* h_ei = (const int*)d_in[2];   // [2,E]: src row0, dst row1
  const int* t_ei = (const int*)d_in[3];
  const int* b_ei = (const int*)d_in[4];   // row0: h-node idx, row1: t-node idx
  const int* h_batch = (const int*)d_in[5];
  const int* t_batch = (const int*)d_in[6];
  const int* h_src = h_ei, *h_dst = h_ei + N_EDGES;
  const int* t_src = t_ei, *t_dst = t_ei + N_EDGES;
  const int* b_h = b_ei, *b_t = b_ei + N_EDGES;

  // params pytree flattened (dict keys in sorted order) starting at d_in[7]:
  // W0, b0, blocks[4]x29, n0_b, n0_w, net_norms[4]x{b,w}
  auto P = [&](int i) -> const float* { return (const float*)d_in[7 + i]; };
  const float* W0 = P(0); const float* b0 = P(1);
  const float* n0_b = P(118); const float* n0_w = P(119);
  // per-tconv leaf order (sorted): Wk,Wq,Ws,Wv,bk,bq,bs,bv
  auto TC = [&](int base) -> TCP {
    TCP t; t.Wk = P(base + 0); t.Wq = P(base + 1); t.Ws = P(base + 2); t.Wv = P(base + 3);
    t.bk = P(base + 4); t.bq = P(base + 5); t.bs = P(base + 6); t.bv = P(base + 7);
    return t;
  };

  // Workspace carve-up
  float* w = (float*)d_ws;
  const size_t NN = N_NODES;
  float* xh  = w; w += NN * 128;
  float* xh2 = w; w += NN * 128;
  float* xt  = w; w += NN * 128;
  float* xt2 = w; w += NN * 128;
  float* qb  = w; w += NN * 128;
  float* kb  = w; w += NN * 128;
  float* vb  = w; w += NN * 128;
  float* elog = w; w += (size_t)N_EDGES * 2;
  float* mb  = w; w += NN * 2;
  float* db  = w; w += NN * 2;
  float* gsum = w; w += N_G;
  float* gss  = w; w += N_G;
  float* cnth = w; w += N_G;
  float* cntt = w; w += N_G;
  float* s1  = w; w += NN;
  float* s2  = w; w += NN;
  float* agg = w; w += NN;
  float* gm  = w; w += N_G;
  float* gden = w; w += N_G;

  float* out = (float*)d_out;                 // [2][G, N_BLK, 128]
  float* out_h = out;
  float* out_t = out + (size_t)N_G * N_BLK * 128;
  launch_fill(stream, out, 0.0f, out_size);

  // Per-graph node counts (used by every graph-LN)
  launch_fill(stream, cnth, 0.0f, N_G);
  launch_fill(stream, cntt, 0.0f, N_G);
  count_nodes<<<(N_NODES + 255) / 256, 256, 0, stream>>>(h_batch, cnth, N_NODES);
  count_nodes<<<(N_NODES + 255) / 256, 256, 0, stream>>>(t_batch, cntt, N_NODES);

  // Input layer: relu(gln(x @ W0 + b0)), 64ch, stored contiguous (ld=64) in xh/xt
  launch_gemm(stream, h_x_in, W0, b0, xh, N_NODES, 64, 64, 64, 0);
  launch_gemm(stream, t_x_in, W0, b0, xt, N_NODES, 64, 64, 64, 0);
  run_gln(stream, xh, h_batch, cnth, n0_w, n0_b, N_NODES, 64, /*relu*/1, gsum, gss);
  run_gln(stream, xt, t_batch, cntt, n0_w, n0_b, N_NODES, 64, /*relu*/1, gsum, gss);

  int din = 64;
  for (int i = 0; i < N_BLK; ++i) {
    const int bb = 2 + 29 * i;
    TCP feat = TC(bb), inter = TC(bb + 8), intra = TC(bb + 16);
    const float* norm_b = P(bb + 24); const float* norm_w = P(bb + 25);
    const float* Wrel = P(bb + 26); const float* Wroot = P(bb + 27); const float* brel = P(bb + 28);
    const float* nn_b = P(120 + 2 * i); const float* nn_w = P(121 + 2 * i);

    // feat tconv (din -> 128), write to xh2/xt2
    run_tconv(stream, xh, xh, din, h_src, h_dst, N_NODES, N_NODES, feat, 64,
              xh2, 128, 0, qb, kb, vb, elog, mb, db);
    run_tconv(stream, xt, xt, din, t_src, t_dst, N_NODES, N_NODES, feat, 64,
              xt2, 128, 0, qb, kb, vb, elog, mb, db);
    // gln + fused elu -> he/te in place (pre-elu value not needed afterwards)
    run_gln(stream, xh2, h_batch, cnth, norm_w, norm_b, N_NODES, 128, /*elu*/2, gsum, gss);
    run_gln(stream, xt2, t_batch, cntt, norm_w, norm_b, N_NODES, 128, /*elu*/2, gsum, gss);

    // intra (cols 0..63) and inter (cols 64..127), concatenated into xh/xt
    run_tconv(stream, xh2, xh2, 128, h_src, h_dst, N_NODES, N_NODES, intra, 32,
              xh, 128, 0, qb, kb, vb, elog, mb, db);
    run_tconv(stream, xt2, xh2, 128, b_t, b_h, N_NODES, N_NODES, inter, 32,
              xh, 128, 64, qb, kb, vb, elog, mb, db);   // h_inter: src=te, dst=he
    run_tconv(stream, xt2, xt2, 128, t_src, t_dst, N_NODES, N_NODES, intra, 32,
              xt, 128, 0, qb, kb, vb, elog, mb, db);
    run_tconv(stream, xh2, xt2, 128, b_h, b_t, N_NODES, N_NODES, inter, 32,
              xt, 128, 64, qb, kb, vb, elog, mb, db);   // t_inter: src=he, dst=te

    // SAG pooled reps (pre net-norm) -> d_out
    run_sag(stream, xh, h_src, h_dst, h_batch, Wrel, Wroot, brel, out_h,
            N_NODES, i, s1, s2, agg, gm, gden);
    run_sag(stream, xt, t_src, t_dst, t_batch, Wrel, Wroot, brel, out_t,
            N_NODES, i, s1, s2, agg, gm, gden);

    // net norm + elu, in place; xh/xt become next block input
    run_gln(stream, xh, h_batch, cnth, nn_w, nn_b, N_NODES, 128, /*elu*/2, gsum, gss);
    run_gln(stream, xt, t_batch, cntt, nn_w, nn_b, N_NODES, 128, /*elu*/2, gsum, gss);
    din = 128;
  }
}
